// SwinBlock__29334626632486
// MI455X (gfx1250) — matrix operand order repaired and checked by reading, not compiled, not run on hardware
//
#include <hip/hip_runtime.h>
#include <hip/hip_bf16.h>
#include <math.h>

typedef __attribute__((ext_vector_type(16))) _Float16 v16h;
typedef __attribute__((ext_vector_type(8)))  _Float16 v8h;
typedef __attribute__((ext_vector_type(8)))  float    v8f;

// ---------------- WMMA helpers (gfx1250, wave32) ----------------

__device__ __forceinline__ v8f wmma_f16(v16h a, v16h b, v8f c) {
  return __builtin_amdgcn_wmma_f32_16x16x32_f16(false, a, false, b, (short)0, c,
                                                false, false);
}

// combine two 16-byte vector loads into one 32-byte fragment
__device__ __forceinline__ v16h ld2(const _Float16* p0, const _Float16* p1) {
  v8h lo = *(const v8h*)p0;
  v8h hi = *(const v8h*)p1;
  v16h r;
#pragma unroll
  for (int i = 0; i < 8; ++i) { r[i] = lo[i]; r[8 + i] = hi[i]; }
  return r;
}

// A fragment: 16x32 f16, row-major source, leading dim lda.
// Lanes 0-15: row=lane,    K = {0..15}
// Lanes16-31: row=lane-16, K = {16..31}
__device__ __forceinline__ v16h ld_a_frag(const _Float16* base, int lda, int lane) {
  const int row = lane & 15;
  const int kb  = (lane & 16) ? 16 : 0;
  const _Float16* p = base + row * lda + kb;
  return ld2(p, p + 8);
}

// B fragment where logical B[k][n] = src[n*ld + k]  (transposed source, k contiguous)
__device__ __forceinline__ v16h ld_bt_frag(const _Float16* base, int ld, int lane) {
  const int col = lane & 15;
  const int kb  = (lane & 16) ? 16 : 0;
  const _Float16* p = base + col * ld + kb;
  return ld2(p, p + 8);
}

// B fragment from pre-swizzled weights: lane's 16 halves stored contiguously.
// tile order: (ct * K32 + ks) ; within tile: 32 lanes x 16 halves
__device__ __forceinline__ v16h ld_b_swz(const _Float16* wsw, int ct, int ks,
                                         int K32, int lane) {
  const _Float16* p = wsw + (((size_t)(ct * K32 + ks) * 32 + lane) << 4);
  return ld2(p, p + 8);
}

// ---------------- weight f32 -> f16 precast + fragment swizzle ----------------
// regions (halves): q(16384) k(16384) v(16384) proj(16384) fc1(65536) fc2(65536) ds(131072)
__device__ __forceinline__ size_t swz_idx(int k, int n, int K) {
  int ct = n >> 4, cl = n & 15;
  int ks = k >> 5, ki = k & 31;
  int lane = cl + (ki & 16);
  int i = ki & 15;
  return (((size_t)(ct * (K >> 5) + ks) * 32 + lane) << 4) + i;
}

__global__ void cvt_w_kernel(const float* __restrict__ qw, const float* __restrict__ kw,
                             const float* __restrict__ vw, const float* __restrict__ pw,
                             const float* __restrict__ f1, const float* __restrict__ f2,
                             const float* __restrict__ ds, _Float16* __restrict__ o) {
  int i = blockIdx.x * 256 + threadIdx.x;
  if (i >= 327680) return;
  const float* src;
  int K, N, off, loc;
  if (i < 16384)       { src = qw; K = 128; N = 128; off = 0;      loc = i; }
  else if (i < 32768)  { src = kw; K = 128; N = 128; off = 16384;  loc = i - 16384; }
  else if (i < 49152)  { src = vw; K = 128; N = 128; off = 32768;  loc = i - 32768; }
  else if (i < 65536)  { src = pw; K = 128; N = 128; off = 49152;  loc = i - 49152; }
  else if (i < 131072) { src = f1; K = 128; N = 512; off = 65536;  loc = i - 65536; }
  else if (i < 196608) { src = f2; K = 512; N = 128; off = 131072; loc = i - 131072; }
  else                 { src = ds; K = 512; N = 256; off = 196608; loc = i - 196608; }
  int k = loc / N, n = loc - k * N;
  o[off + swz_idx(k, n, K)] = (_Float16)src[loc];
}

// ---------------- fused window attention ----------------
// one block per window (B*NW = 4096), 256 threads = 8 waves
__global__ __launch_bounds__(256) void swin_attn_kernel(
    const float* __restrict__ xin, const float* __restrict__ ln1_g,
    const float* __restrict__ ln1_b,
    const _Float16* __restrict__ qwh, const float* __restrict__ q_b,
    const _Float16* __restrict__ kwh, const float* __restrict__ k_b,
    const _Float16* __restrict__ vwh, const float* __restrict__ v_b,
    const _Float16* __restrict__ pwh, const float* __restrict__ proj_b,
    const float* __restrict__ bias_table, float* __restrict__ h1) {
  __shared__ __align__(32) char smem[134048];
  _Float16* Hls  = (_Float16*)(smem + 0);       // 64x128 h, later ctx
  _Float16* Qls  = (_Float16*)(smem + 16384);   // 64x128 (row=pos, col=channel)
  _Float16* Kls  = (_Float16*)(smem + 32768);   // 64x128 (row=pos, col=channel)
  _Float16* Vls  = (_Float16*)(smem + 49152);   // V TRANSPOSED: [128][64] = [channel][pos]
  float*    Sls  = (float*)   (smem + 65536);   // 4x64x64 scores f32
  _Float16* Pls  = (_Float16*)(smem + 16384);   // probs f16, aliases Q+K (dead)
  float*    BTls = (float*)   (smem + 131072);  // 169*4 rel-bias
  int*      TOKls= (int*)     (smem + 133792);  // 64 token indices

  const int w = blockIdx.x;
  const int b = w >> 6;
  const int nw = w & 63;
  const int wh = nw >> 3, ww = nw & 7;
  const int t = threadIdx.x;
  const int lane = t & 31, wave = t >> 5;

  // ---- phase 0: rel-bias -> LDS ; gather + roll + LN1 -> Hls (rows 49..63 zero)
  for (int i = t; i < 676; i += 256) BTls[i] = bias_table[i];
  if (t < 64) {
    if (t < 49) {
      int ay = t / 7, ax = t - 7 * ay;
      int ry = wh * 7 + ay, rx = ww * 7 + ax;
      int y = ry + 3; if (y >= 56) y -= 56;   // roll(h, -3) gather
      int x = rx + 3; if (x >= 56) x -= 56;
      int tok = b * 3136 + y * 56 + x;
      TOKls[t] = tok;
      const float4* s4 = (const float4*)(xin + (size_t)tok * 128);
      float mu = 0.f;
      for (int c = 0; c < 32; ++c) { float4 u = s4[c]; mu += u.x + u.y + u.z + u.w; }
      mu *= (1.f / 128.f);
      float var = 0.f;
      for (int c = 0; c < 32; ++c) {
        float4 u = s4[c];
        float d0 = u.x - mu, d1 = u.y - mu, d2 = u.z - mu, d3 = u.w - mu;
        var += d0 * d0 + d1 * d1 + d2 * d2 + d3 * d3;
      }
      var *= (1.f / 128.f);
      float rs = rsqrtf(var + 1e-5f);
      for (int c = 0; c < 32; ++c) {
        float4 u = s4[c];
        Hls[t * 128 + 4*c + 0] = (_Float16)((u.x - mu) * rs * ln1_g[4*c+0] + ln1_b[4*c+0]);
        Hls[t * 128 + 4*c + 1] = (_Float16)((u.y - mu) * rs * ln1_g[4*c+1] + ln1_b[4*c+1]);
        Hls[t * 128 + 4*c + 2] = (_Float16)((u.z - mu) * rs * ln1_g[4*c+2] + ln1_b[4*c+2]);
        Hls[t * 128 + 4*c + 3] = (_Float16)((u.w - mu) * rs * ln1_g[4*c+3] + ln1_b[4*c+3]);
      }
    } else {
      TOKls[t] = 0;
      for (int c = 0; c < 128; ++c) Hls[t * 128 + c] = (_Float16)0.f;
    }
  }
  __syncthreads();

  // ---- phase 1: Q,K,V = h @ W + b   (96 tiles of 16x16, K=128)
  for (int tile = wave; tile < 96; tile += 8) {
    int m = tile >> 5;
    int rem = tile & 31;
    int r = rem >> 3, c = rem & 7;
    const _Float16* wgt = (m == 0) ? qwh : (m == 1) ? kwh : vwh;
    const float*    bia = (m == 0) ? q_b : (m == 1) ? k_b : v_b;
    v8f acc = {};
#pragma unroll
    for (int ks = 0; ks < 4; ++ks) {
      v16h a  = ld_a_frag(Hls + r * 16 * 128 + ks * 32, 128, lane);
      v16h bf = ld_b_swz(wgt, c, ks, 4, lane);
      acc = wmma_f16(a, bf, acc);
    }
    int col = c * 16 + (lane & 15);
    int rb  = r * 16 + ((lane & 16) ? 8 : 0);
    float bv = bia[col];
    if (m == 2) {
#pragma unroll
      for (int v = 0; v < 8; ++v)
        Vls[col * 64 + (rb + v)] = (_Float16)(acc[v] + bv);   // transposed store
    } else {
      _Float16* dst = (m == 0) ? Qls : Kls;
#pragma unroll
      for (int v = 0; v < 8; ++v)
        dst[(rb + v) * 128 + col] = (_Float16)(acc[v] + bv);
    }
  }
  __syncthreads();

  // ---- phase 2: scores = Q K^T / sqrt(HD) + rel_bias + shift_mask (64 tiles, K=32)
  const float inv_sqrt_hd = 0.17677669529663687f;
  for (int tile = wave; tile < 64; tile += 8) {
    int head = tile >> 4;
    int r = (tile >> 2) & 3, c = tile & 3;
    v16h a  = ld_a_frag(Qls + r * 16 * 128 + head * 32, 128, lane);
    v16h bf = ld_bt_frag(Kls + c * 16 * 128 + head * 32, 128, lane);
    v8f acc = {};
    acc = wmma_f16(a, bf, acc);
    int kk = c * 16 + (lane & 15);
    int qb = r * 16 + ((lane & 16) ? 8 : 0);
#pragma unroll
    for (int v = 0; v < 8; ++v) {
      int q = qb + v;
      float s;
      if (kk < 49 && q < 49) {
        int qy = q / 7,  qx = q - 7 * qy;
        int ky = kk / 7, kx = kk - 7 * ky;
        int ri = (qy - ky + 6) * 13 + (qx - kx + 6);
        float bias = BTls[ri * 4 + head];
        int rqy = wh * 7 + qy, rqx = ww * 7 + qx;
        int rky = wh * 7 + ky, rkx = ww * 7 + kx;
        int lq = 3 * ((rqy < 49) ? 0 : ((rqy < 53) ? 1 : 2)) +
                     ((rqx < 49) ? 0 : ((rqx < 53) ? 1 : 2));
        int lk = 3 * ((rky < 49) ? 0 : ((rky < 53) ? 1 : 2)) +
                     ((rkx < 49) ? 0 : ((rkx < 53) ? 1 : 2));
        float mask = (lq != lk) ? -100.0f : 0.0f;
        s = acc[v] * inv_sqrt_hd + bias + mask;
      } else {
        s = -1e30f;  // padded key/query
      }
      Sls[(head * 64 + q) * 64 + kk] = s;
    }
  }
  __syncthreads();

  // ---- phase 3: softmax (one thread per (head,row)) -> Pls f16
  {
    int head = t >> 6, q = t & 63;
    _Float16* prow = Pls + (head * 64 + q) * 64;
    if (q < 49) {
      const float* srow = Sls + (head * 64 + q) * 64;
      float mx = -1e30f;
      for (int k2 = 0; k2 < 64; ++k2) mx = fmaxf(mx, srow[k2]);
      float sum = 0.f;
      for (int k2 = 0; k2 < 64; ++k2) sum += __expf(srow[k2] - mx);
      float rinv = 1.f / sum;
      for (int k2 = 0; k2 < 64; ++k2)
        prow[k2] = (_Float16)(__expf(srow[k2] - mx) * rinv);
    } else {
      for (int k2 = 0; k2 < 64; ++k2) prow[k2] = (_Float16)0.f;
    }
  }
  __syncthreads();

  // ---- phase 4: ctx = P @ V per head (32 tiles, K=64) -> Hls (h dead)
  for (int tile = wave; tile < 32; tile += 8) {
    int head = tile >> 3;
    int r = (tile >> 1) & 3, c = tile & 1;
    v8f acc = {};
#pragma unroll
    for (int ks = 0; ks < 2; ++ks) {
      v16h a  = ld_a_frag(Pls + head * 4096 + r * 16 * 64 + ks * 32, 64, lane);
      // B[k=pos][n=ch] = Vt[ch][pos] : transposed source, contiguous in k
      v16h bf = ld_bt_frag(Vls + (head * 32 + c * 16) * 64 + ks * 32, 64, lane);
      acc = wmma_f16(a, bf, acc);
    }
    int col = head * 32 + c * 16 + (lane & 15);
    int rb  = r * 16 + ((lane & 16) ? 8 : 0);
#pragma unroll
    for (int v = 0; v < 8; ++v)
      Hls[(rb + v) * 128 + col] = (_Float16)acc[v];
  }
  __syncthreads();

  // ---- phase 5: out = ctx @ proj_w + proj_b + shortcut -> h1 (scatter, inverse roll)
  for (int tile = wave; tile < 32; tile += 8) {
    int r = tile >> 3, c = tile & 7;
    v8f acc = {};
#pragma unroll
    for (int ks = 0; ks < 4; ++ks) {
      v16h a  = ld_a_frag(Hls + r * 16 * 128 + ks * 32, 128, lane);
      v16h bf = ld_b_swz(pwh, c, ks, 4, lane);
      acc = wmma_f16(a, bf, acc);
    }
    int col = c * 16 + (lane & 15);
    int rb  = r * 16 + ((lane & 16) ? 8 : 0);
    float bv = proj_b[col];
#pragma unroll
    for (int v = 0; v < 8; ++v) {
      int row = rb + v;
      if (row < 49) {
        size_t idx = (size_t)TOKls[row] * 128 + col;
        h1[idx] = acc[v] + bv + xin[idx];
      }
    }
  }
}

// ---------------- fused MLP: LN2 -> fc1 -> GELU -> fc2 -> +residual (in-place h1)
__global__ __launch_bounds__(256) void swin_mlp_kernel(
    const float* __restrict__ ln2_g, const float* __restrict__ ln2_b,
    const _Float16* __restrict__ f1wh, const float* __restrict__ fc1_b,
    const _Float16* __restrict__ f2wh, const float* __restrict__ fc2_b,
    float* __restrict__ h1) {
  __shared__ __align__(32) char smem[16384 + 65536];
  _Float16* Lls = (_Float16*)smem;            // 64x128 ln2(h1)
  _Float16* Yls = (_Float16*)(smem + 16384);  // 64x512 gelu(fc1)
  const int t = threadIdx.x, lane = t & 31, wave = t >> 5;
  const size_t row0 = (size_t)blockIdx.x * 64;

  if (t < 64) {
    const float4* s4 = (const float4*)(h1 + (row0 + t) * 128);
    float mu = 0.f;
    for (int c = 0; c < 32; ++c) { float4 u = s4[c]; mu += u.x + u.y + u.z + u.w; }
    mu *= (1.f / 128.f);
    float var = 0.f;
    for (int c = 0; c < 32; ++c) {
      float4 u = s4[c];
      float d0 = u.x - mu, d1 = u.y - mu, d2 = u.z - mu, d3 = u.w - mu;
      var += d0 * d0 + d1 * d1 + d2 * d2 + d3 * d3;
    }
    var *= (1.f / 128.f);
    float rs = rsqrtf(var + 1e-5f);
    for (int c = 0; c < 32; ++c) {
      float4 u = s4[c];
      Lls[t * 128 + 4*c + 0] = (_Float16)((u.x - mu) * rs * ln2_g[4*c+0] + ln2_b[4*c+0]);
      Lls[t * 128 + 4*c + 1] = (_Float16)((u.y - mu) * rs * ln2_g[4*c+1] + ln2_b[4*c+1]);
      Lls[t * 128 + 4*c + 2] = (_Float16)((u.z - mu) * rs * ln2_g[4*c+2] + ln2_b[4*c+2]);
      Lls[t * 128 + 4*c + 3] = (_Float16)((u.w - mu) * rs * ln2_g[4*c+3] + ln2_b[4*c+3]);
    }
  }
  __syncthreads();

  // fc1 + exact GELU : 4x32 tiles, K=128
  for (int tile = wave; tile < 128; tile += 8) {
    int r = tile >> 5, c = tile & 31;
    v8f acc = {};
#pragma unroll
    for (int ks = 0; ks < 4; ++ks) {
      v16h a  = ld_a_frag(Lls + r * 16 * 128 + ks * 32, 128, lane);
      v16h bf = ld_b_swz(f1wh, c, ks, 4, lane);
      acc = wmma_f16(a, bf, acc);
    }
    int col = c * 16 + (lane & 15);
    int rb  = r * 16 + ((lane & 16) ? 8 : 0);
    float bv = fc1_b[col];
#pragma unroll
    for (int v = 0; v < 8; ++v) {
      float xg = acc[v] + bv;
      float g = 0.5f * xg * (1.0f + erff(xg * 0.7071067811865476f));
      Yls[(rb + v) * 512 + col] = (_Float16)g;
    }
  }
  __syncthreads();

  // fc2 + residual (in place): 4x8 tiles, K=512
  for (int tile = wave; tile < 32; tile += 8) {
    int r = tile >> 3, c = tile & 7;
    v8f acc = {};
#pragma unroll
    for (int ks = 0; ks < 16; ++ks) {
      v16h a  = ld_a_frag(Yls + r * 16 * 512 + ks * 32, 512, lane);
      v16h bf = ld_b_swz(f2wh, c, ks, 16, lane);
      acc = wmma_f16(a, bf, acc);
    }
    int col = c * 16 + (lane & 15);
    int rb  = r * 16 + ((lane & 16) ? 8 : 0);
    float bv = fc2_b[col];
#pragma unroll
    for (int v = 0; v < 8; ++v) {
      size_t idx = (row0 + rb + v) * 128 + col;
      h1[idx] = acc[v] + bv + h1[idx];
    }
  }
}

// ---------------- patch-merge downsample: gather 2x2 -> LN -> @ ds_red_w
__global__ __launch_bounds__(256) void swin_ds_kernel(
    const float* __restrict__ g, const float* __restrict__ bb,
    const _Float16* __restrict__ dswh, const float* __restrict__ h2,
    float* __restrict__ out) {
  __shared__ __align__(32) char smem[65536];
  _Float16* Cls = (_Float16*)smem;  // 64x512
  const int t = threadIdx.x, lane = t & 31, wave = t >> 5;
  const size_t row0 = (size_t)blockIdx.x * 64;

  if (t < 64) {
    size_t orow = row0 + t;
    int b = (int)(orow / 784);
    int p = (int)(orow % 784);
    int i = p / 28, j = p - 28 * i;
    size_t t00 = (size_t)b * 3136 + (size_t)(2 * i) * 56 + 2 * j;
    const float4* s0 = (const float4*)(h2 + t00 * 128);         // (2i,   2j)
    const float4* s1 = (const float4*)(h2 + (t00 + 56) * 128);  // (2i+1, 2j)
    const float4* s2 = (const float4*)(h2 + (t00 + 1) * 128);   // (2i,   2j+1)
    const float4* s3 = (const float4*)(h2 + (t00 + 57) * 128);  // (2i+1, 2j+1)
    float mu = 0.f;
    for (int c = 0; c < 32; ++c) {
      float4 a = s0[c], d = s1[c], e = s2[c], f = s3[c];
      mu += a.x + a.y + a.z + a.w + d.x + d.y + d.z + d.w +
            e.x + e.y + e.z + e.w + f.x + f.y + f.z + f.w;
    }
    mu *= (1.f / 512.f);
    float var = 0.f;
    for (int c = 0; c < 32; ++c) {
      float4 a = s0[c], d = s1[c], e = s2[c], f = s3[c];
      float q0;
      q0 = a.x - mu; var += q0 * q0;  q0 = a.y - mu; var += q0 * q0;
      q0 = a.z - mu; var += q0 * q0;  q0 = a.w - mu; var += q0 * q0;
      q0 = d.x - mu; var += q0 * q0;  q0 = d.y - mu; var += q0 * q0;
      q0 = d.z - mu; var += q0 * q0;  q0 = d.w - mu; var += q0 * q0;
      q0 = e.x - mu; var += q0 * q0;  q0 = e.y - mu; var += q0 * q0;
      q0 = e.z - mu; var += q0 * q0;  q0 = e.w - mu; var += q0 * q0;
      q0 = f.x - mu; var += q0 * q0;  q0 = f.y - mu; var += q0 * q0;
      q0 = f.z - mu; var += q0 * q0;  q0 = f.w - mu; var += q0 * q0;
    }
    var *= (1.f / 512.f);
    float rs = rsqrtf(var + 1e-5f);
    _Float16* dst = Cls + t * 512;
    for (int c = 0; c < 32; ++c) {
      float4 a = s0[c], d = s1[c], e = s2[c], f = s3[c];
      int c4 = 4 * c;
      dst[c4+0]     = (_Float16)((a.x-mu)*rs*g[c4+0]     + bb[c4+0]);
      dst[c4+1]     = (_Float16)((a.y-mu)*rs*g[c4+1]     + bb[c4+1]);
      dst[c4+2]     = (_Float16)((a.z-mu)*rs*g[c4+2]     + bb[c4+2]);
      dst[c4+3]     = (_Float16)((a.w-mu)*rs*g[c4+3]     + bb[c4+3]);
      dst[128+c4+0] = (_Float16)((d.x-mu)*rs*g[128+c4+0] + bb[128+c4+0]);
      dst[128+c4+1] = (_Float16)((d.y-mu)*rs*g[128+c4+1] + bb[128+c4+1]);
      dst[128+c4+2] = (_Float16)((d.z-mu)*rs*g[128+c4+2] + bb[128+c4+2]);
      dst[128+c4+3] = (_Float16)((d.w-mu)*rs*g[128+c4+3] + bb[128+c4+3]);
      dst[256+c4+0] = (_Float16)((e.x-mu)*rs*g[256+c4+0] + bb[256+c4+0]);
      dst[256+c4+1] = (_Float16)((e.y-mu)*rs*g[256+c4+1] + bb[256+c4+1]);
      dst[256+c4+2] = (_Float16)((e.z-mu)*rs*g[256+c4+2] + bb[256+c4+2]);
      dst[256+c4+3] = (_Float16)((e.w-mu)*rs*g[256+c4+3] + bb[256+c4+3]);
      dst[384+c4+0] = (_Float16)((f.x-mu)*rs*g[384+c4+0] + bb[384+c4+0]);
      dst[384+c4+1] = (_Float16)((f.y-mu)*rs*g[384+c4+1] + bb[384+c4+1]);
      dst[384+c4+2] = (_Float16)((f.z-mu)*rs*g[384+c4+2] + bb[384+c4+2]);
      dst[384+c4+3] = (_Float16)((f.w-mu)*rs*g[384+c4+3] + bb[384+c4+3]);
    }
  }
  __syncthreads();

  // 4x16 tiles, K=512
  for (int tile = wave; tile < 64; tile += 8) {
    int r = tile >> 4, c = tile & 15;
    v8f acc = {};
#pragma unroll
    for (int ks = 0; ks < 16; ++ks) {
      v16h a  = ld_a_frag(Cls + r * 16 * 512 + ks * 32, 512, lane);
      v16h bf = ld_b_swz(dswh, c, ks, 16, lane);
      acc = wmma_f16(a, bf, acc);
    }
    int col = c * 16 + (lane & 15);
    int rb  = r * 16 + ((lane & 16) ? 8 : 0);
#pragma unroll
    for (int v = 0; v < 8; ++v)
      out[(row0 + rb + v) * 256 + col] = acc[v];
  }
}

extern "C" void kernel_launch(void* const* d_in, const int* in_sizes, int n_in,
                              void* d_out, int out_size, void* d_ws, size_t ws_size,
                              hipStream_t stream) {
  (void)in_sizes; (void)n_in; (void)out_size; (void)ws_size;
  const float* x      = (const float*)d_in[0];
  const float* ln1_g  = (const float*)d_in[1];
  const float* ln1_b  = (const float*)d_in[2];
  const float* q_w    = (const float*)d_in[3];
  const float* q_b    = (const float*)d_in[4];
  const float* k_w    = (const float*)d_in[5];
  const float* k_b    = (const float*)d_in[6];
  const float* v_w    = (const float*)d_in[7];
  const float* v_b    = (const float*)d_in[8];
  const float* proj_w = (const float*)d_in[9];
  const float* proj_b = (const float*)d_in[10];
  const float* relb   = (const float*)d_in[11];
  const float* ln2_g  = (const float*)d_in[12];
  const float* ln2_b  = (const float*)d_in[13];
  const float* fc1_w  = (const float*)d_in[14];
  const float* fc1_b  = (const float*)d_in[15];
  const float* fc2_w  = (const float*)d_in[16];
  const float* fc2_b  = (const float*)d_in[17];
  const float* dsg    = (const float*)d_in[18];
  const float* dsb    = (const float*)d_in[19];
  const float* dsw    = (const float*)d_in[20];

  _Float16* wh = (_Float16*)d_ws;               // 655360 bytes of swizzled f16 weights
  float* h1 = (float*)((char*)d_ws + 655360);   // 200704 x 128 f32 residual stream

  cvt_w_kernel<<<1280, 256, 0, stream>>>(q_w, k_w, v_w, proj_w, fc1_w, fc2_w, dsw, wh);
  swin_attn_kernel<<<4096, 256, 0, stream>>>(
      x, ln1_g, ln1_b,
      wh + 0,     q_b,
      wh + 16384, k_b,
      wh + 32768, v_b,
      wh + 49152, proj_b, relb, h1);
  swin_mlp_kernel<<<3136, 256, 0, stream>>>(ln2_g, ln2_b,
                                            wh + 65536, fc1_b,
                                            wh + 131072, fc2_b, h1);
  swin_ds_kernel<<<784, 256, 0, stream>>>(dsg, dsb, wh + 196608, h1, (float*)d_out);
}
